// InfiniteAttention_15822659518688
// MI455X (gfx1250) — matrix operand order repaired and checked by reading, not compiled, not run on hardware
//
#include <hip/hip_runtime.h>

typedef __bf16 bf16;
typedef __attribute__((ext_vector_type(16))) __bf16 v16bf;
typedef __attribute__((ext_vector_type(8)))  float v8f;
typedef __attribute__((ext_vector_type(4)))  unsigned v4u;
typedef __attribute__((ext_vector_type(8)))  int v8i;
typedef __attribute__((ext_vector_type(4)))  int v4i;

#define SEQ 2048
#define DIM 2048
#define NH  32
#define NKV 8
#define HD  64
#define QD  (NH * HD)    // 2048
#define KVD (NKV * HD)   // 512

#if __has_builtin(__builtin_amdgcn_tensor_load_to_lds) && \
    __has_builtin(__builtin_amdgcn_s_wait_tensorcnt)
#define HAS_TDM 1
#else
#define HAS_TDM 0
#endif

union Frag {
  v16bf v;
  unsigned u[8];
  bf16 h[16];
};

__device__ __forceinline__ v8f vzero8() {
  v8f z = {0.f, 0.f, 0.f, 0.f, 0.f, 0.f, 0.f, 0.f};
  return z;
}

// A fragment: 16x32 bf16, row-major LDS tile (stride ld halves).
// Lane L<16: M=L, K in {0..7, 16..23}; lane L>=16: M=L-16, K in {8..15, 24..31}.
__device__ __forceinline__ v16bf load_a_frag(const bf16* __restrict__ base, int ld,
                                             int m0, int k0, int lane) {
  Frag f;
  int m = m0 + (lane & 15);
  int kh = (lane >> 4) * 8;
  const bf16* row = base + m * ld + k0;
#pragma unroll
  for (int i = 0; i < 8; ++i) {
    int k = (i < 4) ? (kh + 2 * i) : (16 + kh + 2 * (i - 4));
    f.u[i] = *(const unsigned*)(row + k);
  }
  return f.v;
}

// Same as load_a_frag but applies elu(x)+1 elementwise (compressive-memory sig_q).
__device__ __forceinline__ v16bf load_a_frag_sig(const bf16* __restrict__ base, int ld,
                                                 int m0, int k0, int lane) {
  Frag f;
  int m = m0 + (lane & 15);
  int kh = (lane >> 4) * 8;
  const bf16* row = base + m * ld + k0;
#pragma unroll
  for (int i = 0; i < 16; ++i) {
    int pr = i >> 1;
    int k = ((pr < 4) ? (kh + 2 * pr) : (16 + kh + 2 * (pr - 4))) + (i & 1);
    float x = (float)row[k];
    f.h[i] = (bf16)(x > 0.f ? x + 1.f : __expf(x));
  }
  return f.v;
}

// B fragment: 32x16 bf16 from LDS holding B^T row-major [N][K] (stride ld halves).
// Lane L<16: N=L, K in {0..7, 16..23}; lane L>=16: N=L-16, K in {8..15, 24..31}.
__device__ __forceinline__ v16bf load_b_frag(const bf16* __restrict__ bt, int ld,
                                             int n0, int k0, int lane) {
  Frag f;
  int n = n0 + (lane & 15);
  int kb = (lane >> 4) * 8;
  const bf16* row = bt + n * ld + k0;
#pragma unroll
  for (int i = 0; i < 8; ++i) f.u[i] = *(const unsigned*)(row + ((i < 4) ? (kb + 2 * i) : (16 + kb + 2 * (i - 4))));
  return f.v;
}

#define WMMA_BF16(A, B, C) \
  __builtin_amdgcn_wmma_f32_16x16x32_bf16(false, (A), false, (B), (short)0, (C), false, false)

// TDM: DMA a [rows x 32] bf16 tile (row-major, row stride `stride` elements)
// from global into LDS. Issued by one wave, tracked with TENSORcnt.
// This toolchain's builtin takes 6 args: (u32x4 g0, i32x8 g1, i32x4, i32x4, i32x8, i32 cpol).
__device__ __forceinline__ void tdm_load_tile_a(const bf16* gsrc, bf16* lds_dst,
                                                int rows, int stride) {
#if HAS_TDM
  unsigned long long ga = (unsigned long long)(size_t)(const void*)gsrc;
  unsigned ldsa = (unsigned)(unsigned long long)(size_t)(void*)lds_dst;
  // D# group 0: count=1 | lds_addr | global_addr[56:0] | type=2 (bits 127:126)
  v4u g0;
  g0[0] = 1u;
  g0[1] = ldsa;
  g0[2] = (unsigned)(ga & 0xffffffffull);
  g0[3] = (unsigned)((ga >> 32) & 0x01ffffffull) | 0x80000000u;
  // D# group 1: mask=0, data_size=1 (2B), dims/tile/stride per ISA bit layout
  unsigned td0 = (unsigned)stride;   // tensor_dim0 (elements per row)
  unsigned td1 = (unsigned)SEQ;      // tensor_dim1 (rows available)
  unsigned tile0 = 32u;              // tile x (contiguous K chunk)
  unsigned tile1 = (unsigned)rows;   // tile y
  unsigned long long s0 = (unsigned long long)stride;  // tensor_dim0_stride
  v8i g1;
  g1[0] = (int)(1u << 16);                                   // data_size=1 -> 2 bytes
  g1[1] = (int)((td0 & 0xffffu) << 16);                      // tensor_dim0[15:0]
  g1[2] = (int)(((td0 >> 16) & 0xffffu) | ((td1 & 0xffffu) << 16));
  g1[3] = (int)(((td1 >> 16) & 0xffffu) | (tile0 << 16));    // tile_dim0
  g1[4] = (int)(tile1 & 0xffffu);                            // tile_dim1, tile_dim2=0
  g1[5] = (int)(s0 & 0xffffffffull);                         // stride0 lo
  g1[6] = (int)((s0 >> 32) & 0xffffull);                     // stride0 hi | stride1 lo=0
  g1[7] = 0;
  v4i zz4 = {0, 0, 0, 0};
  v8i zz8 = {0, 0, 0, 0, 0, 0, 0, 0};
  __builtin_amdgcn_tensor_load_to_lds(g0, g1, zz4, zz4, zz8, 0);
#else
  (void)gsrc; (void)lds_dst; (void)rows; (void)stride;
#endif
}

// ---------------------------------------------------------------- convert
__global__ __launch_bounds__(256) void cvt_bf16_kernel(const float* __restrict__ in,
                                                       bf16* __restrict__ out, int n) {
  int i = (blockIdx.x * 256 + threadIdx.x) * 4;
  if (i + 3 < n) {
    float4 v = *(const float4*)(in + i);
    out[i + 0] = (bf16)v.x;
    out[i + 1] = (bf16)v.y;
    out[i + 2] = (bf16)v.z;
    out[i + 3] = (bf16)v.w;
  }
}

// ---------------------------------------------------------------- GEMM
// C[M,N] = A[M,K] @ B[K,N], A/B bf16 row-major. Block tile 128x64, 8 waves of 32x32.
// A tiles staged by the Tensor Data Mover (double buffered); B tiles staged
// transposed by the waves. MODE 0/1: Q/K + RoPE -> [heads][SEQ][HD] bf16;
// MODE 2: V -> [NKV][SEQ][HD] bf16; MODE 3: plain f32 row-major [M,N].
template <int MODE>
__global__ __launch_bounds__(256) void gemm_bf16_kernel(const bf16* __restrict__ A,
                                                        const bf16* __restrict__ B,
                                                        void* __restrict__ Out,
                                                        int K, int N) {
  __shared__ __align__(16) bf16 As[2][128 * 32];
  __shared__ __align__(16) bf16 Bts[2][64 * 32];

  int tid = threadIdx.x;
  int lane = tid & 31;
  int w = tid >> 5;
  int m_blk = blockIdx.y * 128;
  int n_blk = blockIdx.x * 64;
  int wr = (w & 3) * 32;
  int wc = (w >> 2) * 32;

  v8f c00 = vzero8(), c01 = vzero8(), c10 = vzero8(), c11 = vzero8();

  int arow = tid >> 1;
  int acol = (tid & 1) * 16;
  int bk = tid >> 3;
  int bn = (tid & 7) * 8;

  // --- stage slab kk=0 into buffer 0
#if HAS_TDM
  if (tid < 32) tdm_load_tile_a(A + (size_t)m_blk * K, &As[0][0], 128, K);
#else
  {
    const bf16* src = A + (size_t)(m_blk + arow) * K + acol;
    *(uint4*)(As[0] + arow * 32 + acol)     = *(const uint4*)(src);
    *(uint4*)(As[0] + arow * 32 + acol + 8) = *(const uint4*)(src + 8);
  }
#endif
  {
    const bf16* src = B + (size_t)bk * N + n_blk + bn;
    bf16 tmp[8];
    *(uint4*)(tmp) = *(const uint4*)(src);
#pragma unroll
    for (int j = 0; j < 8; ++j) Bts[0][(bn + j) * 32 + bk] = tmp[j];
  }

  for (int kk = 0; kk < K; kk += 32) {
    int buf = (kk >> 5) & 1;
#if HAS_TDM
    if (tid < 32) __builtin_amdgcn_s_wait_tensorcnt(0);
#endif
    __syncthreads();  // slab `buf` ready; previous compute finished

    if (kk + 32 < K) {
      // Prefetch next slab into the other buffer while we compute.
#if HAS_TDM
      if (tid < 32)
        tdm_load_tile_a(A + (size_t)m_blk * K + kk + 32, &As[buf ^ 1][0], 128, K);
#else
      {
        const bf16* src = A + (size_t)(m_blk + arow) * K + kk + 32 + acol;
        *(uint4*)(As[buf ^ 1] + arow * 32 + acol)     = *(const uint4*)(src);
        *(uint4*)(As[buf ^ 1] + arow * 32 + acol + 8) = *(const uint4*)(src + 8);
      }
#endif
      {
        const bf16* src = B + (size_t)(kk + 32 + bk) * N + n_blk + bn;
        bf16 tmp[8];
        *(uint4*)(tmp) = *(const uint4*)(src);
#pragma unroll
        for (int j = 0; j < 8; ++j) Bts[buf ^ 1][(bn + j) * 32 + bk] = tmp[j];
      }
      __builtin_prefetch(B + (size_t)(kk + 64 + bk) * N + n_blk + bn, 0, 3);
    }

    v16bf a0 = load_a_frag(As[buf], 32, wr, 0, lane);
    v16bf a1 = load_a_frag(As[buf], 32, wr + 16, 0, lane);
    v16bf b0 = load_b_frag(Bts[buf], 32, wc, 0, lane);
    v16bf b1 = load_b_frag(Bts[buf], 32, wc + 16, 0, lane);
    c00 = WMMA_BF16(a0, b0, c00);
    c01 = WMMA_BF16(a0, b1, c01);
    c10 = WMMA_BF16(a1, b0, c10);
    c11 = WMMA_BF16(a1, b1, c11);
  }

  // Epilogue: C layout -> VGPR r holds row r + 8*(lane>=16), col = lane&15
  int half8 = (lane >> 4) << 3;
  int ncl = lane & 15;
#pragma unroll
  for (int q = 0; q < 4; ++q) {
    v8f c = (q == 0) ? c00 : (q == 1) ? c01 : (q == 2) ? c10 : c11;
    int mbase = m_blk + wr + ((q >> 1) ? 16 : 0) + half8;
    int nc = n_blk + wc + ((q & 1) ? 16 : 0) + ncl;
#pragma unroll
    for (int r = 0; r < 8; ++r) {
      float v = c[r];
      int m = mbase + r;
      if (MODE == 3) {
        ((float*)Out)[(size_t)m * N + nc] = v;
      } else if (MODE == 2) {
        int hh = nc >> 6, d = nc & 63;
        ((bf16*)Out)[((size_t)hh * SEQ + m) * HD + d] = (bf16)v;
      } else {
        float p = __shfl_xor(v, 1, 32);  // partner column d^1
        int hh = nc >> 6, d = nc & 63;
        float ang = (float)m * __powf(10000.0f, -(float)(d & ~1) * (1.0f / 64.0f));
        float cs, sn;
        __sincosf(ang, &cs, &sn);
        float res = (d & 1) ? fmaf(p, sn, v * cs) : fmaf(v, cs, -(p * sn));
        ((bf16*)Out)[((size_t)hh * SEQ + m) * HD + d] = (bf16)res;
      }
    }
  }
}

// ---------------------------------------------------------------- attention
// Grid: (SEQ/64, NH). Block: 128 threads = 4 waves; wave w owns query rows w*16..+15.
__global__ __launch_bounds__(128) void attn_kernel(const bf16* __restrict__ qw,
                                                   const bf16* __restrict__ kw,
                                                   const bf16* __restrict__ vw,
                                                   const float* __restrict__ Mw,
                                                   const float* __restrict__ zw,
                                                   const float* __restrict__ betaw,
                                                   bf16* __restrict__ outw) {
  __shared__ __align__(16) bf16 Qs[64 * 64];
  __shared__ __align__(16) bf16 Ks[64 * 64];
  __shared__ __align__(16) bf16 Vts[64 * 64];   // V transposed: [d][key]
  __shared__ __align__(16) bf16 Ps[4][16 * 64]; // per-wave P tile
  __shared__ __align__(16) bf16 Mts[64 * 64];   // M transposed: [e][d]
  __shared__ float dens[64];

  int tid = threadIdx.x;
  int lane = tid & 31;
  int w = tid >> 5;
  int h = blockIdx.y;
  int ib = blockIdx.x;
  int kvh = h >> 2;  // GQA n_rep = 4

  const bf16* qh = qw + (size_t)h * SEQ * HD;
  const bf16* kh = kw + (size_t)kvh * SEQ * HD;
  const bf16* vh = vw + (size_t)kvh * SEQ * HD;

  int srow = tid >> 1;
  int scol = (tid & 1) * 32;

  // Stage Q tile (rows ib*64..+63) and Mt (once)
  {
    const bf16* src = qh + (size_t)(ib * 64 + srow) * HD + scol;
#pragma unroll
    for (int j = 0; j < 4; ++j)
      *(uint4*)(Qs + srow * 64 + scol + 8 * j) = *(const uint4*)(src + 8 * j);
    const float* Mh = Mw + (size_t)h * HD * HD;
    for (int idx = tid; idx < 64 * 64; idx += 128) {
      int d = idx >> 6, e = idx & 63;
      Mts[e * 64 + d] = (bf16)Mh[idx];
    }
  }
  __syncthreads();

  float mrun[8], lrun[8], alpha[8];
#pragma unroll
  for (int r = 0; r < 8; ++r) { mrun[r] = -1e30f; lrun[r] = 0.f; }
  v8f o0 = vzero8(), o1 = vzero8(), o2 = vzero8(), o3 = vzero8();

  int half8 = (lane >> 4) << 3;
  int kcl = lane & 15;
  int qrow0 = w * 16;

  for (int jb = 0; jb <= ib; ++jb) {
    // Stage K [key][d] (natural == B^T for Q.K^T) and V transposed [d][key]
    {
      const bf16* ksrc = kh + (size_t)(jb * 64 + srow) * HD + scol;
#pragma unroll
      for (int j = 0; j < 4; ++j)
        *(uint4*)(Ks + srow * 64 + scol + 8 * j) = *(const uint4*)(ksrc + 8 * j);
      const bf16* vsrc = vh + (size_t)(jb * 64 + srow) * HD + scol;
      bf16 tmp[32];
#pragma unroll
      for (int j = 0; j < 4; ++j) *(uint4*)(tmp + 8 * j) = *(const uint4*)(vsrc + 8 * j);
#pragma unroll
      for (int j = 0; j < 32; ++j) Vts[(scol + j) * 64 + srow] = tmp[j];
    }
    __syncthreads();

    // S = Q @ K^T (16 q-rows x 64 keys per wave); hoist all fragments first
    v8f s0 = vzero8(), s1 = vzero8(), s2 = vzero8(), s3 = vzero8();
#pragma unroll
    for (int k2 = 0; k2 < 64; k2 += 32) {
      v16bf aq  = load_a_frag(Qs, 64, qrow0, k2, lane);
      v16bf bk0 = load_b_frag(Ks, 64, 0,  k2, lane);
      v16bf bk1 = load_b_frag(Ks, 64, 16, k2, lane);
      v16bf bk2 = load_b_frag(Ks, 64, 32, k2, lane);
      v16bf bk3 = load_b_frag(Ks, 64, 48, k2, lane);
      s0 = WMMA_BF16(aq, bk0, s0);
      s1 = WMMA_BF16(aq, bk1, s1);
      s2 = WMMA_BF16(aq, bk2, s2);
      s3 = WMMA_BF16(aq, bk3, s3);
    }

    // Online softmax per query row r (row data lives across 16 lanes x 4 tiles)
#pragma unroll
    for (int r = 0; r < 8; ++r) {
      float sv0 = s0[r] * 0.125f, sv1 = s1[r] * 0.125f;
      float sv2 = s2[r] * 0.125f, sv3 = s3[r] * 0.125f;
      int q = ib * 64 + qrow0 + half8 + r;
      int kb = jb * 64 + kcl;
      if (kb + 0  > q) sv0 = -1e30f;
      if (kb + 16 > q) sv1 = -1e30f;
      if (kb + 32 > q) sv2 = -1e30f;
      if (kb + 48 > q) sv3 = -1e30f;
      float mx = fmaxf(fmaxf(sv0, sv1), fmaxf(sv2, sv3));
#pragma unroll
      for (int off = 1; off < 16; off <<= 1) mx = fmaxf(mx, __shfl_xor(mx, off, 32));
      float mnew = fmaxf(mrun[r], mx);
      float a = __expf(mrun[r] - mnew);
      sv0 = __expf(sv0 - mnew); sv1 = __expf(sv1 - mnew);
      sv2 = __expf(sv2 - mnew); sv3 = __expf(sv3 - mnew);
      float rs = sv0 + sv1 + sv2 + sv3;
#pragma unroll
      for (int off = 1; off < 16; off <<= 1) rs += __shfl_xor(rs, off, 32);
      lrun[r] = lrun[r] * a + rs;
      mrun[r] = mnew;
      alpha[r] = a;
      int prow = r + half8;
      Ps[w][prow * 64 + 0  + kcl] = (bf16)sv0;
      Ps[w][prow * 64 + 16 + kcl] = (bf16)sv1;
      Ps[w][prow * 64 + 32 + kcl] = (bf16)sv2;
      Ps[w][prow * 64 + 48 + kcl] = (bf16)sv3;
    }

    // Rescale running O, then O += P @ V
#pragma unroll
    for (int r = 0; r < 8; ++r) {
      o0[r] *= alpha[r]; o1[r] *= alpha[r]; o2[r] *= alpha[r]; o3[r] *= alpha[r];
    }
#pragma unroll
    for (int k2 = 0; k2 < 64; k2 += 32) {
      v16bf ap  = load_a_frag(Ps[w], 64, 0, k2, lane);
      v16bf bv0 = load_b_frag(Vts, 64, 0,  k2, lane);
      v16bf bv1 = load_b_frag(Vts, 64, 16, k2, lane);
      v16bf bv2 = load_b_frag(Vts, 64, 32, k2, lane);
      v16bf bv3 = load_b_frag(Vts, 64, 48, k2, lane);
      o0 = WMMA_BF16(ap, bv0, o0);
      o1 = WMMA_BF16(ap, bv1, o1);
      o2 = WMMA_BF16(ap, bv2, o2);
      o3 = WMMA_BF16(ap, bv3, o3);
    }
    __syncthreads();
  }

  // Compressive memory read: num = (elu(q)+1) @ M
  v8f n0 = vzero8(), n1 = vzero8(), n2 = vzero8(), n3 = vzero8();
#pragma unroll
  for (int k2 = 0; k2 < 64; k2 += 32) {
    v16bf as  = load_a_frag_sig(Qs, 64, qrow0, k2, lane);
    v16bf bm0 = load_b_frag(Mts, 64, 0,  k2, lane);
    v16bf bm1 = load_b_frag(Mts, 64, 16, k2, lane);
    v16bf bm2 = load_b_frag(Mts, 64, 32, k2, lane);
    v16bf bm3 = load_b_frag(Mts, 64, 48, k2, lane);
    n0 = WMMA_BF16(as, bm0, n0);
    n1 = WMMA_BF16(as, bm1, n1);
    n2 = WMMA_BF16(as, bm2, n2);
    n3 = WMMA_BF16(as, bm3, n3);
  }
  // den = sig_q . z + eps (one row per thread, f32)
  if (tid < 64) {
    const float* zh = zw + (size_t)h * HD;
    float s = 0.f;
    for (int d = 0; d < 64; ++d) {
      float x = (float)Qs[tid * 64 + d];
      float sg = x > 0.f ? x + 1.f : __expf(x);
      s = fmaf(sg, zh[d], s);
    }
    dens[tid] = s + 1e-8f;
  }
  __syncthreads();

  float g = 1.f / (1.f + __expf(-betaw[h]));
#pragma unroll
  for (int t = 0; t < 4; ++t) {
    v8f oc = (t == 0) ? o0 : (t == 1) ? o1 : (t == 2) ? o2 : o3;
    v8f nc = (t == 0) ? n0 : (t == 1) ? n1 : (t == 2) ? n2 : n3;
    int d = t * 16 + kcl;
#pragma unroll
    for (int r = 0; r < 8; ++r) {
      int mloc = qrow0 + half8 + r;
      float lo = oc[r] / lrun[r];
      float am = nc[r] / dens[mloc];
      float res = fmaf(g, am, (1.f - g) * lo);
      int s = ib * 64 + mloc;
      outw[(size_t)s * QD + h * HD + d] = (bf16)res;
    }
  }
}

// ---------------------------------------------------------------- launch
extern "C" void kernel_launch(void* const* d_in, const int* in_sizes, int n_in,
                              void* d_out, int out_size, void* d_ws, size_t ws_size,
                              hipStream_t stream) {
  (void)in_sizes; (void)n_in; (void)out_size; (void)ws_size;
  const float* x    = (const float*)d_in[0];
  const float* wq   = (const float*)d_in[1];
  const float* wk   = (const float*)d_in[2];
  const float* wv   = (const float*)d_in[3];
  const float* wo   = (const float*)d_in[4];
  const float* beta = (const float*)d_in[5];
  const float* M    = (const float*)d_in[6];
  const float* z    = (const float*)d_in[7];

  char* ws = (char*)d_ws;
  bf16* xbf  = (bf16*)(ws + 0);                         // 8 MiB (reused for attn out)
  bf16* wqbf = (bf16*)(ws + (size_t)8  * 1048576);      // 8 MiB
  bf16* wkbf = (bf16*)(ws + (size_t)16 * 1048576);      // 2 MiB
  bf16* wvbf = (bf16*)(ws + (size_t)18 * 1048576);      // 2 MiB
  bf16* wobf = (bf16*)(ws + (size_t)20 * 1048576);      // 8 MiB
  bf16* qws  = (bf16*)(ws + (size_t)28 * 1048576);      // 8 MiB  [NH][SEQ][HD]
  bf16* kws  = (bf16*)(ws + (size_t)36 * 1048576);      // 2 MiB  [NKV][SEQ][HD]
  bf16* vws  = (bf16*)(ws + (size_t)38 * 1048576);      // 2 MiB  [NKV][SEQ][HD]
  bf16* attn = xbf;                                     // x dead after QKV GEMMs

  // f32 -> bf16 conversions
  cvt_bf16_kernel<<<(SEQ * DIM) / 1024, 256, 0, stream>>>(x, xbf, SEQ * DIM);
  cvt_bf16_kernel<<<(DIM * QD) / 1024, 256, 0, stream>>>(wq, wqbf, DIM * QD);
  cvt_bf16_kernel<<<(DIM * KVD) / 1024, 256, 0, stream>>>(wk, wkbf, DIM * KVD);
  cvt_bf16_kernel<<<(DIM * KVD) / 1024, 256, 0, stream>>>(wv, wvbf, DIM * KVD);
  cvt_bf16_kernel<<<(QD * DIM) / 1024, 256, 0, stream>>>(wo, wobf, QD * DIM);

  // QKV projections (+RoPE for Q/K) -> head-major layouts
  gemm_bf16_kernel<0><<<dim3(QD / 64, SEQ / 128), 256, 0, stream>>>(xbf, wqbf, qws, DIM, QD);
  gemm_bf16_kernel<1><<<dim3(KVD / 64, SEQ / 128), 256, 0, stream>>>(xbf, wkbf, kws, DIM, KVD);
  gemm_bf16_kernel<2><<<dim3(KVD / 64, SEQ / 128), 256, 0, stream>>>(xbf, wvbf, vws, DIM, KVD);

  // Fused flash attention + compressive memory + gate -> attn [SEQ][QD] bf16
  attn_kernel<<<dim3(SEQ / 64, NH), 128, 0, stream>>>(qws, kws, vws, M, z, beta, attn);

  // Output projection -> f32 d_out
  gemm_bf16_kernel<3><<<dim3(DIM / 64, SEQ / 128), 256, 0, stream>>>(attn, wobf, d_out, QD, DIM);
}
